// MambaBlock3D_59554016526637
// MI455X (gfx1250) — matrix operand order repaired and checked by reading, not compiled, not run on hardware
//
#include <hip/hip_runtime.h>
#include <hip/hip_bf16.h>

typedef __attribute__((ext_vector_type(16))) __bf16 v16bf;
typedef __attribute__((ext_vector_type(8)))  float  v8f;

#define D_MODEL 128
#define D_INNER 128
#define D_STATE 16
#define DT_RANK 8
#define BATCH   2
#define SEQL    8192            // 8*32*32
#define MTOT    (BATCH * SEQL)  // 16384
#define KTILES  4               // K=128 -> 4 x 32
#define NPAD2   48              // x_proj N=40 padded to 48

// ---------- helpers ----------

__device__ __forceinline__ unsigned short f2bf(float f) {
    union { float f; unsigned int u; } v; v.f = f;
    unsigned int u = v.u;
    u += 0x7FFFu + ((u >> 16) & 1u);   // round-to-nearest-even
    return (unsigned short)(u >> 16);
}

// A-fragment scatter offset (16-bit A matrix 16x32, wave32 layout, ISA 7.12.2)
// element (m, k) -> lane = (m&15) | (((k>>3)&1)<<4) ; half i = (k&7) | (((k>>4)&1)<<3)
__device__ __forceinline__ size_t afrag_offset(int m, int k, int ktiles) {
    int kk   = k & 31;
    int lane = (m & 15) | (((kk >> 3) & 1) << 4);
    int i    = (kk & 7) | (((kk >> 4) & 1) << 3);
    return ((size_t)(m >> 4) * ktiles + (size_t)(k >> 5)) * 512 + lane * 16 + i;
}

// ---------- weight packing: W[n, k] (row-major NxK) -> B fragments (KxN tiles) ----------
// B tile layout (matches A table order): k = kt*32 + (i&7) + 8*(lane>>4) + 16*(i>>3)
// ; n = nt*16 + (lane&15)
__global__ void pack_b_kernel(const float* __restrict__ W, unsigned short* __restrict__ Bf,
                              int Nrows, int K, int ntiles, int ktiles) {
    int idx   = blockIdx.x * blockDim.x + threadIdx.x;
    int total = ktiles * ntiles * 512;
    if (idx >= total) return;
    int i    = idx & 15;
    int lane = (idx >> 4) & 31;
    int tile = idx >> 9;
    int nt   = tile % ntiles;
    int kt   = tile / ntiles;
    int n    = nt * 16 + (lane & 15);
    int k    = kt * 32 + (i & 7) + ((lane >> 4) << 3) + ((i >> 3) << 4);
    float v  = (n < Nrows && k < K) ? W[(size_t)n * K + k] : 0.f;
    Bf[idx]  = f2bf(v);
}

// ---------- BN + transpose, write A-fragment-packed bf16 ----------
__global__ void bn_pack_kernel(const float* __restrict__ x,
                               const float* __restrict__ g,  const float* __restrict__ be,
                               const float* __restrict__ mu, const float* __restrict__ var,
                               unsigned short* __restrict__ xfA) {
    int idx = blockIdx.x * blockDim.x + threadIdx.x;   // b*C*L + c*L + l  (l fastest -> coalesced read)
    if (idx >= BATCH * D_MODEL * SEQL) return;
    int l = idx & (SEQL - 1);
    int c = (idx >> 13) & (D_MODEL - 1);
    int b = idx >> 20;
    float inv = g[c] * rsqrtf(var[c] + 1e-5f);
    float v   = (x[idx] - mu[c]) * inv + be[c];
    int m = b * SEQL + l;
    xfA[afrag_offset(m, c, KTILES)] = f2bf(v);
}

// ---------- generic WMMA GEMM: A packed frags [mtiles][ktiles], B packed frags [ktiles][ntiles] ----------
template <int NT>
__global__ __launch_bounds__(256)
void wmma_gemm_kernel(const unsigned short* __restrict__ Af,
                      const unsigned short* __restrict__ Bf,
                      float* __restrict__ C,
                      int mtiles, int ntiles, int ktiles, int ldc) {
    int gwave = (blockIdx.x * blockDim.x + threadIdx.x) >> 5;
    int lane  = threadIdx.x & 31;
    int ngroups = ntiles / NT;
    int mt = gwave / ngroups;
    int ng = gwave - mt * ngroups;
    if (mt >= mtiles) return;

    v8f acc[NT];
    #pragma unroll
    for (int j = 0; j < NT; ++j)
        #pragma unroll
        for (int r = 0; r < 8; ++r) acc[j][r] = 0.f;

    const unsigned short* Abase = Af + (size_t)mt * ktiles * 512 + lane * 16;
    for (int kt = 0; kt < ktiles; ++kt) {
        v16bf a = *(const v16bf*)(Abase + (size_t)kt * 512);
        #pragma unroll
        for (int j = 0; j < NT; ++j) {
            const unsigned short* Bt =
                Bf + ((size_t)kt * ntiles + (ng * NT + j)) * 512 + lane * 16;
            v16bf b = *(const v16bf*)Bt;
            acc[j] = __builtin_amdgcn_wmma_f32_16x16x32_bf16(
                         false, a, false, b, (short)0, acc[j], false, false);
        }
    }
    // C/D layout: VGPR r -> M = r (lanes 0-15) / r+8 (lanes 16-31); N = lane&15
    int mrow0 = mt * 16 + ((lane >> 4) << 3);
    int ncol  = lane & 15;
    #pragma unroll
    for (int j = 0; j < NT; ++j) {
        int n = (ng * NT + j) * 16 + ncol;
        #pragma unroll
        for (int r = 0; r < 8; ++r)
            C[(size_t)(mrow0 + r) * ldc + n] = acc[j][r];
    }
}

// ---------- causal depthwise conv(4) + bias + SiLU; emits f32 + packed bf16 ----------
__global__ void conv_silu_kernel(const float* __restrict__ xz,   // [M, 256], cols 0..127 = xs_pre
                                 const float* __restrict__ w,    // [128,1,4]
                                 const float* __restrict__ bias,
                                 float* __restrict__ xs,         // [M, 128] f32
                                 unsigned short* __restrict__ xsA) {
    int idx = blockIdx.x * blockDim.x + threadIdx.x;  // m*128 + c (c fastest -> coalesced)
    if (idx >= MTOT * D_INNER) return;
    int c = idx & 127;
    int m = idx >> 7;
    int l = m & (SEQL - 1);
    float acc = bias[c];
    #pragma unroll
    for (int t = 0; t < 4; ++t) {
        int ll = l - 3 + t;
        if (ll >= 0) acc += w[c * 4 + t] * xz[(size_t)(m - 3 + t) * 256 + c];
    }
    float s = acc / (1.f + __expf(-acc));  // SiLU
    xs[idx] = s;
    xsA[afrag_offset(m, c, KTILES)] = f2bf(s);
}

// ---------- dt = softplus(dt_in * Wdt^T + b), K=8 (too small for WMMA) ----------
__global__ void dt_kernel(const float* __restrict__ xdbl,  // [M, 48]
                          const float* __restrict__ Wdt,   // [128, 8]
                          const float* __restrict__ bdt,
                          float* __restrict__ dt) {
    int idx = blockIdx.x * blockDim.x + threadIdx.x;  // m*128 + d
    if (idx >= MTOT * D_INNER) return;
    int d = idx & 127;
    int m = idx >> 7;
    float acc = bdt[d];
    #pragma unroll
    for (int r = 0; r < DT_RANK; ++r)
        acc += xdbl[(size_t)m * NPAD2 + r] * Wdt[d * DT_RANK + r];
    dt[idx] = (acc > 20.f) ? acc : log1pf(__expf(acc));
}

// ---------- selective scan: one 16-lane half-wave per (b,d); lane n owns state h[n] ----------
__global__ void scan_kernel(const float* __restrict__ dt,    // [M,128]
                            const float* __restrict__ xs,    // [M,128]
                            const float* __restrict__ xdbl,  // [M,48]: B=8..23, C=24..39
                            const float* __restrict__ xz,    // [M,256]: z = 128..255
                            const float* __restrict__ A_log, // [128,16]
                            const float* __restrict__ Dp,    // [128]
                            unsigned short* __restrict__ yA) {
    int tid  = blockIdx.x * blockDim.x + threadIdx.x;
    int seq  = tid >> 4;          // (b,d) sequence id
    int n    = tid & 15;          // state index
    if (seq >= BATCH * D_INNER) return;
    int d = seq & 127;
    int b = seq >> 7;
    float A  = -__expf(A_log[d * D_STATE + n]);
    float Dv = Dp[d];
    float h  = 0.f;
    size_t mbase = (size_t)b * SEQL;
    for (int l = 0; l < SEQL; ++l) {
        size_t m   = mbase + l;
        float dtv  = dt[m * 128 + d];
        float xv   = xs[m * 128 + d];
        float Bc   = xdbl[m * NPAD2 + DT_RANK + n];
        float Cc   = xdbl[m * NPAD2 + DT_RANK + D_STATE + n];
        h = __expf(dtv * A) * h + dtv * Bc * xv;
        float acc = h * Cc;
        acc += __shfl_xor(acc, 1);
        acc += __shfl_xor(acc, 2);
        acc += __shfl_xor(acc, 4);
        acc += __shfl_xor(acc, 8);
        if (n == 0) {
            float zv = xz[m * 256 + 128 + d];
            float y  = (acc + xv * Dv) * (zv / (1.f + __expf(-zv)));
            yA[afrag_offset((int)m, d, KTILES)] = f2bf(y);
        }
    }
}

// ---------- out = relu(yproj^T + x), back to (B,C,L) ----------
__global__ void epilogue_kernel(const float* __restrict__ yproj,  // [M,128]
                                const float* __restrict__ x,
                                float* __restrict__ out) {
    int idx = blockIdx.x * blockDim.x + threadIdx.x;  // b*C*L + c*L + l
    if (idx >= BATCH * D_MODEL * SEQL) return;
    int l = idx & (SEQL - 1);
    int c = (idx >> 13) & (D_MODEL - 1);
    int b = idx >> 20;
    size_t m = (size_t)b * SEQL + l;
    float v = yproj[m * 128 + c] + x[idx];
    out[idx] = fmaxf(v, 0.f);
}

// ---------- launcher ----------
extern "C" void kernel_launch(void* const* d_in, const int* in_sizes, int n_in,
                              void* d_out, int out_size, void* d_ws, size_t ws_size,
                              hipStream_t stream) {
    (void)in_sizes; (void)n_in; (void)out_size; (void)ws_size;
    const float* x       = (const float*)d_in[0];
    const float* bn_g    = (const float*)d_in[1];
    const float* bn_b    = (const float*)d_in[2];
    const float* bn_mu   = (const float*)d_in[3];
    const float* bn_var  = (const float*)d_in[4];
    const float* in_w    = (const float*)d_in[5];   // [256,128]
    const float* conv_w  = (const float*)d_in[6];   // [128,1,4]
    const float* conv_b  = (const float*)d_in[7];
    const float* xproj_w = (const float*)d_in[8];   // [40,128]
    const float* dt_w    = (const float*)d_in[9];   // [128,8]
    const float* dt_b    = (const float*)d_in[10];
    const float* A_log   = (const float*)d_in[11];  // [128,16]
    const float* D_param = (const float*)d_in[12];
    const float* out_w   = (const float*)d_in[13];  // [128,128]
    float* out = (float*)d_out;

    // workspace carve-up (all 256B aligned)
    char* ws = (char*)d_ws;
    size_t off = 0;
    auto carve = [&](size_t bytes) { char* p = ws + off; off = (off + bytes + 255) & ~(size_t)255; return p; };
    unsigned short* xfA  = (unsigned short*)carve((size_t)MTOT * 128 * 2);   // packed A for GEMM1
    unsigned short* w1f  = (unsigned short*)carve((size_t)KTILES * 16 * 512 * 2);
    unsigned short* w2f  = (unsigned short*)carve((size_t)KTILES * 3  * 512 * 2);
    unsigned short* w3f  = (unsigned short*)carve((size_t)KTILES * 8  * 512 * 2);
    float*          xz   = (float*)         carve((size_t)MTOT * 256 * 4);
    float*          xs   = (float*)         carve((size_t)MTOT * 128 * 4);
    unsigned short* xsA  = (unsigned short*)carve((size_t)MTOT * 128 * 2);
    float*          xdbl = (float*)         carve((size_t)MTOT * NPAD2 * 4);
    float*          dt   = (float*)         carve((size_t)MTOT * 128 * 4);
    unsigned short* yA   = (unsigned short*)carve((size_t)MTOT * 128 * 2);
    float*          yprj = (float*)         carve((size_t)MTOT * 128 * 4);

    const int TPB = 256;
    // 1) pack weights into B-fragment order
    pack_b_kernel<<<(KTILES*16*512 + TPB-1)/TPB, TPB, 0, stream>>>(in_w,    w1f, 256, 128, 16, KTILES);
    pack_b_kernel<<<(KTILES*3 *512 + TPB-1)/TPB, TPB, 0, stream>>>(xproj_w, w2f,  40, 128,  3, KTILES);
    pack_b_kernel<<<(KTILES*8 *512 + TPB-1)/TPB, TPB, 0, stream>>>(out_w,   w3f, 128, 128,  8, KTILES);
    // 2) BN + transpose -> packed A
    bn_pack_kernel<<<(BATCH*D_MODEL*SEQL + TPB-1)/TPB, TPB, 0, stream>>>(x, bn_g, bn_b, bn_mu, bn_var, xfA);
    // 3) in_proj: [16384,128] x [128,256]
    { int waves = (MTOT/16) * (16/4);
      wmma_gemm_kernel<4><<<(waves*32 + TPB-1)/TPB, TPB, 0, stream>>>(xfA, w1f, xz, MTOT/16, 16, KTILES, 256); }
    // 4) causal conv + SiLU
    conv_silu_kernel<<<(MTOT*D_INNER + TPB-1)/TPB, TPB, 0, stream>>>(xz, conv_w, conv_b, xs, xsA);
    // 5) x_proj: [16384,128] x [128,48(pad 40)]
    { int waves = (MTOT/16) * (3/3);
      wmma_gemm_kernel<3><<<(waves*32 + TPB-1)/TPB, TPB, 0, stream>>>(xsA, w2f, xdbl, MTOT/16, 3, KTILES, NPAD2); }
    // 6) dt projection + softplus
    dt_kernel<<<(MTOT*D_INNER + TPB-1)/TPB, TPB, 0, stream>>>(xdbl, dt_w, dt_b, dt);
    // 7) selective scan (fused D-skip, silu(z) gate), emits packed A for GEMM3
    { int threads = BATCH * D_INNER * D_STATE;   // 4096
      scan_kernel<<<(threads + TPB-1)/TPB, TPB, 0, stream>>>(dt, xs, xdbl, xz, A_log, D_param, yA); }
    // 8) out_proj: [16384,128] x [128,128]
    { int waves = (MTOT/16) * (8/4);
      wmma_gemm_kernel<4><<<(waves*32 + TPB-1)/TPB, TPB, 0, stream>>>(yA, w3f, yprj, MTOT/16, 8, KTILES, 128); }
    // 9) residual + relu, transpose back to (B,C,D,H,W)
    epilogue_kernel<<<(BATCH*D_MODEL*SEQL + TPB-1)/TPB, TPB, 0, stream>>>(yprj, x, out);
}
